// MuSc_85633057948278
// MI455X (gfx1250) — compile-verified, matrix operand order checked by reading, not run on hardware
//
#include <hip/hip_runtime.h>
#include <hip/hip_bf16.h>

#define PDIM 1369
#define DDIM 1024
#define NREFT 86          // ceil(1369/16)
#define HOUT 518
#define BUFHALVES (16 * 1024 + 32 * 8)   // 16 rows x 1024 + 16B pad per 256 DW

typedef _Float16 v16h __attribute__((ext_vector_type(16)));
typedef _Float16 v8h  __attribute__((ext_vector_type(8)));
typedef float    v8f  __attribute__((ext_vector_type(8)));
typedef unsigned int u32x4 __attribute__((ext_vector_type(4)));
typedef int i32x8 __attribute__((ext_vector_type(8)));
typedef int i32x4 __attribute__((ext_vector_type(4)));

// ---------------------------------------------------------------------------
// 1) Per-(l,b) layernorm statistics over the full (P,D) slice.
// ---------------------------------------------------------------------------
__global__ __launch_bounds__(256) void musc_stats_kernel(
    const float* __restrict__ feat, float* __restrict__ stats)
{
    __shared__ float rs_[256];
    __shared__ float rss[256];
    const int slice = blockIdx.x;                // 0..31  (l*8+b)
    const float* base = feat + (size_t)slice * PDIM * DDIM;
    float s = 0.f, ss = 0.f;
    for (int idx = threadIdx.x; idx < PDIM * DDIM; idx += 256) {
        float v = base[idx];
        s += v; ss += v * v;
    }
    rs_[threadIdx.x] = s; rss[threadIdx.x] = ss;
    __syncthreads();
    for (int k = 128; k > 0; k >>= 1) {
        if (threadIdx.x < k) {
            rs_[threadIdx.x] += rs_[threadIdx.x + k];
            rss[threadIdx.x] += rss[threadIdx.x + k];
        }
        __syncthreads();
    }
    if (threadIdx.x == 0) {
        const float N  = (float)(PDIM * DDIM);
        const float mu = rs_[0] / N;
        const float var = rss[0] / N - mu * mu;
        stats[2 * slice]     = mu;
        stats[2 * slice + 1] = rsqrtf(var + 1e-5f);
    }
}

// ---------------------------------------------------------------------------
// 2) Box filter (zero-padded, /r^2) of the normalized features + L2 norm
//    along D, emitted as f16  [l][b][p][d].
// ---------------------------------------------------------------------------
__global__ __launch_bounds__(256) void musc_prep_kernel(
    const float* __restrict__ feat, const float* __restrict__ stats,
    _Float16* __restrict__ out, int r)
{
    __shared__ float red[256];
    const int p = blockIdx.x, b = blockIdx.y, l = blockIdx.z;
    const int slice = l * 8 + b;
    const float mu = stats[2 * slice];
    const float rs = stats[2 * slice + 1];
    const int ph = p / 37, pw = p % 37, pad = r >> 1;
    const float* base = feat + (size_t)slice * PDIM * DDIM;

    float acc[4] = {0.f, 0.f, 0.f, 0.f};
    int nv = 0;
    for (int dy = -pad; dy <= pad; dy++) {
        const int y = ph + dy;
        if (y < 0 || y >= 37) continue;
        for (int dx = -pad; dx <= pad; dx++) {
            const int x = pw + dx;
            if (x < 0 || x >= 37) continue;
            nv++;
            const float* row = base + (size_t)(y * 37 + x) * DDIM;
#pragma unroll
            for (int k = 0; k < 4; k++) acc[k] += row[threadIdx.x + 256 * k];
        }
    }
    const float inv = 1.0f / (float)(r * r);
    float vals[4]; float ss = 0.f;
#pragma unroll
    for (int k = 0; k < 4; k++) {
        vals[k] = rs * (acc[k] - (float)nv * mu) * inv;  // zero-padded mean
        ss += vals[k] * vals[k];
    }
    red[threadIdx.x] = ss;
    __syncthreads();
    for (int s = 128; s > 0; s >>= 1) {
        if (threadIdx.x < s) red[threadIdx.x] += red[threadIdx.x + s];
        __syncthreads();
    }
    const float rn = rsqrtf(fmaxf(red[0], 1e-24f));
    _Float16* o = out + ((size_t)slice * PDIM + p) * DDIM;
#pragma unroll
    for (int k = 0; k < 4; k++)
        o[threadIdx.x + 256 * k] = (_Float16)(vals[k] * rn);
}

// ---------------------------------------------------------------------------
// 3) MSM core: max-inner-product GEMM via WMMA.
//    Ref tiles staged via TDM with LDS double-buffering; K-loop software-
//    pipelined; per-lane running argmax, one butterfly per ref image.
//    Block = 8 waves x 16 query rows. grid = (11, L=4, B=8).
// ---------------------------------------------------------------------------
__global__ __launch_bounds__(256) void musc_msm_kernel(
    const _Float16* __restrict__ F,      // [L][B][P][D] f16 normalized
    float* __restrict__ scores,          // [B][P] accumulator (x12 pairs)
    float* __restrict__ out_mi,          // [B][L][3][7][P] argmin indices (as float)
    int rr)
{
    __shared__ __align__(16) _Float16 lds_ref[2][BUFHALVES];

    const int tid    = threadIdx.x;
    const int lane   = tid & 31;
    const int wave   = tid >> 5;
    const int laneLo = lane & 15;
    const int laneHi = lane >> 4;
    const int tile = blockIdx.x, l = blockIdx.y, i = blockIdx.z;
    const int qrow0 = tile * 128 + wave * 16;
    const int qm = min(qrow0 + laneLo, PDIM - 1);   // clamp; extra rows discarded
    // A-fragment base (16-bit A 16x32 layout): laneHi selects K sub-chunk
    const _Float16* Q = F + ((size_t)(l * 8 + i)) * PDIM * DDIM
                          + (size_t)qm * DDIM + laneHi * 8;

    const unsigned ldsbase = (unsigned)(size_t)(&lds_ref[0][0]);

    // Issue one TDM load of a 16 x 1024 f16 ref tile into LDS buffer bufp.
    auto issueTDM = [&](int jjp, int np, int bufp) {
        const int jimgp = jjp + (jjp >= i ? 1 : 0);
        const unsigned long long ga = (unsigned long long)(size_t)
            (F + ((size_t)(l * 8 + jimgp)) * PDIM * DDIM + (size_t)np * 16 * DDIM);
        const int rows_rem = PDIM - np * 16;         // OOB rows read as zero
        u32x4 g0;
        g0.x = 1u;                                   // count=1, user descriptor
        g0.y = ldsbase + (unsigned)bufp * (unsigned)(BUFHALVES * 2);
        g0.z = (unsigned)ga;                         // global addr [31:0]
        g0.w = (unsigned)(ga >> 32) | (2u << 30);    // addr[56:32] | type=2
        i32x8 g1;
        g1[0] = (1 << 16)        // data_size = 2 bytes
              | (1 << 20)        // pad_enable
              | (7 << 22)        // pad_interval = 256 DWORDs
              | (3 << 25);       // pad_amount   = 4 DWORDs (16B)
        g1[1] = (DDIM & 0xFFFF) << 16;               // tensor_dim0 lo16
        g1[2] = (rows_rem & 0xFFFF) << 16;           // tensor_dim1 lo16
        g1[3] = (DDIM & 0xFFFF) << 16;               // tile_dim0
        g1[4] = 16;                                  // tile_dim1=16, tile_dim2=0
        g1[5] = DDIM;                                // tensor_dim0_stride lo32
        g1[6] = 0;
        g1[7] = 0;
        i32x4 gz  = {0, 0, 0, 0};
        i32x8 gz8 = {0, 0, 0, 0, 0, 0, 0, 0};
        __builtin_amdgcn_tensor_load_to_lds(g0, g1, gz, gz, gz8, 0);
    };

    float m1[8], m2[8];                  // two smallest dists across ref images
#pragma unroll
    for (int v = 0; v < 8; v++) { m1[v] = 1e30f; m2[v] = 1e30f; }

    float bestv[8]; int besti[8];        // per-lane running argmax (per column)
#pragma unroll
    for (int v = 0; v < 8; v++) { bestv[v] = -1e30f; besti[v] = 0; }

    if (wave == 0) {
        issueTDM(0, 0, 0);
        __builtin_amdgcn_s_wait_tensorcnt(0);
    }
    __syncthreads();

    int jj = 0, n = 0;
    for (int t = 0; t < 7 * NREFT; t++) {
        // Prefetch next tile into the other buffer (overlaps with compute).
        if (wave == 0) {
            int nn = n + 1, jjn = jj;
            if (nn == NREFT) { nn = 0; jjn++; }
            if (jjn < 7) issueTDM(jjn, nn, (t + 1) & 1);
        }

        // B-fragment base: row laneLo; pad-adjusted row stride 1040 halves.
        const _Float16* Brow = &lds_ref[t & 1][0] + laneLo * 1040 + laneHi * 8;

        v8f c = {};
        v8h a0 = *(const v8h*)(Q);
        v8h a1 = *(const v8h*)(Q + 16);
        v8h b0 = *(const v8h*)(Brow);
        v8h b1 = *(const v8h*)(Brow + 16);
#pragma unroll
        for (int kc = 0; kc < DDIM; kc += 32) {
            v16h a = __builtin_shufflevector(a0, a1,
                     0,1,2,3,4,5,6,7,8,9,10,11,12,13,14,15);
            v16h b = __builtin_shufflevector(b0, b1,
                     0,1,2,3,4,5,6,7,8,9,10,11,12,13,14,15);
            if (kc + 32 < DDIM) {                    // software pipeline: k+1
                const int kn = kc + 32;
                const int po = (kn >= 512) ? 8 : 0;  // LDS pad offset (halves)
                a0 = *(const v8h*)(Q + kn);
                a1 = *(const v8h*)(Q + kn + 16);
                b0 = *(const v8h*)(Brow + kn + po);
                b1 = *(const v8h*)(Brow + kn + po + 16);
            }
            c = __builtin_amdgcn_wmma_f32_16x16x32_f16(
                    false, a, false, b, (short)0, c, false, false);
        }

        // Per-lane running max: lane owns column col0 of this tile.
        const int col0 = n * 16 + laneLo;
        const bool okc = (col0 < PDIM);
#pragma unroll
        for (int v = 0; v < 8; v++) {
            const float val = c[v];
            if (okc && val > bestv[v]) { bestv[v] = val; besti[v] = col0; }
        }

        n++;
        if (n == NREFT) {
            // End of ref image jj: merge 16 columns via butterfly, record.
#pragma unroll
            for (int v = 0; v < 8; v++) {
                float val = bestv[v]; int col = besti[v];
#pragma unroll
                for (int m = 1; m < 16; m <<= 1) {
                    const float ov = __shfl_xor(val, m, 32);
                    const int   oc = __shfl_xor(col, m, 32);
                    if (ov > val || (ov == val && oc < col)) { val = ov; col = oc; }
                }
                const int p = qrow0 + laneHi * 8 + v;
                if (laneLo == 0 && p < PDIM) {
                    out_mi[(((((size_t)i * 4 + l) * 3 + rr) * 7 + jj) * PDIM) + p] =
                        (float)col;
                }
                const float dist = sqrtf(fmaxf(2.0f - 2.0f * val, 0.0f));
                if (dist < m1[v]) { m2[v] = m1[v]; m1[v] = dist; }
                else if (dist < m2[v]) { m2[v] = dist; }
                bestv[v] = -1e30f; besti[v] = 0;
            }
            n = 0; jj++;
        }

        if (wave == 0) __builtin_amdgcn_s_wait_tensorcnt(0);
        __syncthreads();
    }

    if (laneLo == 0) {
#pragma unroll
        for (int v = 0; v < 8; v++) {
            const int p = qrow0 + laneHi * 8 + v;
            if (p < PDIM)
                atomicAdd(&scores[(size_t)i * PDIM + p], 0.5f * (m1[v] + m2[v]));
        }
    }
}

// ---------------------------------------------------------------------------
// 4) Final image scores: max over patches, cls-token RSCIN for k=1,2,3.
// ---------------------------------------------------------------------------
__global__ __launch_bounds__(256) void musc_final_kernel(
    const float* __restrict__ cls, const float* __restrict__ scores,
    float* __restrict__ out_final)
{
    __shared__ float red[256];
    __shared__ float simg[8];
    __shared__ float nrm[8];
    __shared__ float sim[8][8];
    const int tid = threadIdx.x;
    for (int b = 0; b < 8; b++) {
        float m = -1e30f;
        for (int p = tid; p < PDIM; p += 256) m = fmaxf(m, scores[b * PDIM + p]);
        red[tid] = m; __syncthreads();
        for (int s = 128; s > 0; s >>= 1) {
            if (tid < s) red[tid] = fmaxf(red[tid], red[tid + s]);
            __syncthreads();
        }
        if (tid == 0) simg[b] = red[0] * (1.0f / 12.0f);
        __syncthreads();
    }
    if (tid < 8) {
        float s = 0.f;
        for (int d = 0; d < 768; d++) { float v = cls[tid * 768 + d]; s += v * v; }
        nrm[tid] = rsqrtf(s);
    }
    __syncthreads();
    if (tid < 64) {
        const int i2 = tid >> 3, j2 = tid & 7;
        float d = 0.f;
        for (int d0 = 0; d0 < 768; d0++) d += cls[i2 * 768 + d0] * cls[j2 * 768 + d0];
        sim[i2][j2] = d * nrm[i2] * nrm[j2];
    }
    __syncthreads();
    if (tid < 8) {
        const int i2 = tid;
        float acc = 0.f;
        for (int k = 1; k <= 3; k++) {
            bool sel[8] = {false, false, false, false, false, false, false, false};
            for (int t = 0; t < k; t++) {
                float best = -1e30f; int bi = 0;
                for (int j2 = 0; j2 < 8; j2++)
                    if (!sel[j2] && sim[i2][j2] > best) { best = sim[i2][j2]; bi = j2; }
                sel[bi] = true;
            }
            float wsum = 0.f, val = 0.f;
            for (int j2 = 0; j2 < 8; j2++) if (sel[j2]) wsum += sim[i2][j2];
            for (int j2 = 0; j2 < 8; j2++) if (sel[j2]) val += sim[i2][j2] * simg[j2];
            acc += val / wsum;
        }
        out_final[i2] = acc * (1.0f / 3.0f);
    }
}

// ---------------------------------------------------------------------------
// 5) Align-corners bilinear upsample 37x37 -> 518x518.
// ---------------------------------------------------------------------------
__global__ __launch_bounds__(256) void musc_pixel_kernel(
    const float* __restrict__ scores, float* __restrict__ out_pixel)
{
    const int idx = blockIdx.x * 256 + threadIdx.x;
    const int b = blockIdx.y;
    if (idx >= HOUT * HOUT) return;
    const int h = idx / HOUT, w = idx % HOUT;
    const float sc = 36.0f / 517.0f;
    const float chf = h * sc, cwf = w * sc;
    int hl = (int)chf; int hh = min(hl + 1, 36); const float fh = chf - (float)hl;
    int wl = (int)cwf; int wh = min(wl + 1, 36); const float fw = cwf - (float)wl;
    const float* S = scores + (size_t)b * PDIM;
    const float v00 = S[hl * 37 + wl], v01 = S[hl * 37 + wh];
    const float v10 = S[hh * 37 + wl], v11 = S[hh * 37 + wh];
    const float v = (1.f - fh) * ((1.f - fw) * v00 + fw * v01)
                  +        fh  * ((1.f - fw) * v10 + fw * v11);
    out_pixel[(size_t)b * HOUT * HOUT + idx] = v * (1.0f / 12.0f);
}

__global__ void musc_zero_kernel(float* __restrict__ p, int n)
{
    const int i = blockIdx.x * 256 + threadIdx.x;
    if (i < n) p[i] = 0.f;
}

// ---------------------------------------------------------------------------
extern "C" void kernel_launch(void* const* d_in, const int* in_sizes, int n_in,
                              void* d_out, int out_size, void* d_ws, size_t ws_size,
                              hipStream_t stream)
{
    const float* features = (const float*)d_in[0];   // (4,8,1369,1024) f32
    const float* cls      = (const float*)d_in[1];   // (8,768) f32

    float* out       = (float*)d_out;
    float* out_final = out;                               // 8
    float* out_pixel = out + 8;                           // 8*518*518
    float* out_mi    = out + 8 + 8 * HOUT * HOUT;         // 8*4*3*7*1369

    char* ws = (char*)d_ws;
    float*    stats   = (float*)ws;                 // 64 floats
    float*    scores  = (float*)(ws + 256);         // 8*1369 floats
    _Float16* featf16 = (_Float16*)(ws + 65536);    // 4*8*1369*1024 halves

    musc_stats_kernel<<<32, 256, 0, stream>>>(features, stats);
    musc_zero_kernel<<<(8 * PDIM + 255) / 256, 256, 0, stream>>>(scores, 8 * PDIM);

    const int rlist[3] = {1, 3, 5};
    for (int rr = 0; rr < 3; rr++) {
        musc_prep_kernel<<<dim3(PDIM, 8, 4), 256, 0, stream>>>(
            features, stats, featf16, rlist[rr]);
        musc_msm_kernel<<<dim3(11, 4, 8), 256, 0, stream>>>(
            featf16, scores, out_mi, rr);
    }

    musc_final_kernel<<<1, 256, 0, stream>>>(cls, scores, out_final);
    musc_pixel_kernel<<<dim3((HOUT * HOUT + 255) / 256, 8), 256, 0, stream>>>(
        scores, out_pixel);
}
